// SubgraphMatching_56573309223903
// MI455X (gfx1250) — compile-verified, matrix operand order checked by reading, not run on hardware
//
#include <hip/hip_runtime.h>
#include <hip/hip_bf16.h>

// ---------------------------------------------------------------------------
// SubgraphMatching on MI455X (gfx1250, wave32, WMMA)
//
// Pipeline (all f32 state, bf16 only inside WMMA A/B operands):
//   enc:   h = relu(X@W1+b1); e = h@W2+b2          -> gemm128_wmma x4
//   deg:   deg[row]+=1 ; dinv = deg>0 ? deg^-0.5:0 -> degree_k, dinv_k
//   gcn l: lin = relu?(x)@Wl+bl                    -> gemm128_wmma
//          acc[row] += lin[col]*dinv[row]*dinv[col]-> zero_f32 + scatter_k
//          (relu fused into next consumer's load)
//   proj:  qp = relu(q)@MW1[:128]; tp = relu(t)@MW1[128:]
//   score: out[i,j] = sum_k relu(qp[i,k]+tp[j,k]+b1[k])*w2[k] + b2
//
// Round-1 -> Round-2 changes:
//   * W staged TRANSPOSED in LDS (Wt[n][k]) so the B fragment is two
//     contiguous 16B chunks per lane -> ds_load_b128 instead of 16x ds_load_u16
//   * A and Wt rows padded by 8 bf16 (row stride 272B = 68 dwords) -> 16B
//     alignment kept for b128 while rows stagger 4 banks -> conflict-free
// ---------------------------------------------------------------------------

typedef __attribute__((ext_vector_type(16))) __bf16 v16bf;
typedef __attribute__((ext_vector_type(8)))  float  v8f;

#define HID 128
#define PAD 8   // bf16 row padding: keeps 16B chunk alignment, staggers banks

// Generic 128-wide GEMM: Out[N x 128] = act(X[N x 128]) @ W[128 x 128] + bias
// One block = 256 threads = 8 waves; block computes a 16-row stripe.
// Wave w owns output columns [16w, 16w+16). K loop = 4 x v_wmma_f32_16x16x32_bf16.
__global__ __launch_bounds__(256)
void gemm128_wmma(const float* __restrict__ X, const float* __restrict__ W,
                  const float* __restrict__ bias, float* __restrict__ Out,
                  int in_relu, int out_relu)
{
    __shared__ __bf16 As[16][HID + PAD];    // activation tile (bf16, K-major)
    __shared__ __bf16 Wt[HID][HID + PAD];   // weight panel TRANSPOSED (n-major)

    const int tid  = threadIdx.x;
    const int row0 = blockIdx.x * 16;

    // Stage W transposed: coalesced global read, strided LDS bf16 store.
    for (int i = tid; i < HID * HID; i += 256) {
        int r = i >> 7, c = i & 127;        // W[r][c], r = k, c = n
        Wt[c][r] = (__bf16)W[i];
    }
    // Stage activation tile, fusing the previous layer's ReLU if requested.
    for (int i = tid; i < 16 * HID; i += 256) {
        int r = i >> 7, c = i & 127;
        float v = X[(row0 + r) * HID + c];
        if (in_relu) v = v > 0.f ? v : 0.f;
        As[r][c] = (__bf16)v;
    }
    __syncthreads();

    const int wave = tid >> 5;
    const int lane = tid & 31;
    const int m    = lane & 15;             // A row (M) in tile
    const int h    = lane >> 4;             // lane half -> K-subblock / M-half
    const int ncol = wave * 16 + (lane & 15);

    const __bf16* __restrict__ arow = &As[m][0];
    const __bf16* __restrict__ brow = &Wt[ncol][0];

    v8f acc = {};
#pragma unroll
    for (int kt = 0; kt < 4; ++kt) {        // K = 128 in 4 steps of 32
        // ISA 7.12.2 16-bit operand layout: vector elems 0..7  <-> k0+0..7,
        //                                   elems 8..15 <-> k0+16..23
        // with k0 = 32*kt + 8*laneHalf. Both chunks are 16B-aligned in LDS.
        const int k0 = kt * 32 + h * 8;
        v16bf a, b;
#pragma unroll
        for (int e = 0; e < 8; ++e) {
            a[e]     = arow[k0 + e];
            a[8 + e] = arow[k0 + 16 + e];
            b[e]     = brow[k0 + e];
            b[8 + e] = brow[k0 + 16 + e];
        }
        acc = __builtin_amdgcn_wmma_f32_16x16x32_bf16(
                  false, a, false, b, (short)0, acc, false, false);
    }

    // D layout: lanes 0-15 -> M = r, lanes 16-31 -> M = 8 + r, N = lane&15.
    const float bv = bias ? bias[ncol] : 0.f;
    const int mb = h * 8;
#pragma unroll
    for (int r = 0; r < 8; ++r) {
        float v = acc[r] + bv;
        if (out_relu) v = v > 0.f ? v : 0.f;
        Out[(row0 + mb + r) * HID + ncol] = v;
    }
}

__global__ void zero_f32(float* __restrict__ p, int n)
{
    int i = blockIdx.x * 256 + threadIdx.x;
    if (i < n) p[i] = 0.f;
}

// deg[row] += 1 per edge (edges stored [2][E]: row = e[i], col = e[E+i])
__global__ void degree_k(const int* __restrict__ edges, float* deg, int E)
{
    int e = blockIdx.x * 256 + threadIdx.x;
    if (e < E) atomicAdd(&deg[edges[e]], 1.0f);
}

__global__ void dinv_k(float* __restrict__ deg, int n)
{
    int i = blockIdx.x * 256 + threadIdx.x;
    if (i < n) {
        float d = deg[i];
        deg[i] = d > 0.f ? rsqrtf(d) : 0.f;
    }
}

// acc[row,:] += lin[col,:] * dinv[row]*dinv[col]; 128 lanes per edge, 2 edges/block.
__global__ __launch_bounds__(256)
void scatter_k(const int* __restrict__ edges, const float* __restrict__ lin,
               const float* __restrict__ dinv, float* __restrict__ acc, int E)
{
    int e = blockIdx.x * 2 + (threadIdx.x >> 7);
    int k = threadIdx.x & 127;
    if (e >= E) return;
    int row = edges[e];
    int col = edges[E + e];
    float norm = dinv[row] * dinv[col];
    atomicAdd(&acc[row * HID + k], lin[col * HID + k] * norm);
}

// out[i,j] = sum_k relu(qp[i,k]+tp[j,k]+b1[k]) * w2[k] + b2
// Block: 16x16 pair tile; qp/tp tiles in LDS padded to stride 132 (bank stagger).
__global__ __launch_bounds__(256)
void pairwise_k(const float* __restrict__ qp, const float* __restrict__ tp,
                const float* __restrict__ b1, const float* __restrict__ w2,
                const float* __restrict__ b2, float* __restrict__ out, int NT)
{
    __shared__ float qs[16][132];
    __shared__ float ts[16][132];
    __shared__ float b1s[HID], w2s[HID];

    const int tid = threadIdx.x;
    const int tj0 = blockIdx.x * 16;
    const int qi0 = blockIdx.y * 16;

    if (tid < HID) { b1s[tid] = b1[tid]; w2s[tid] = w2[tid]; }
    for (int i = tid; i < 16 * HID; i += 256) {
        int r = i >> 7, c = i & 127;
        qs[r][c] = qp[(qi0 + r) * HID + c];
        ts[r][c] = tp[(tj0 + r) * HID + c];
    }
    __syncthreads();

    const int i = tid >> 4;
    const int j = tid & 15;
    float s = 0.f;
#pragma unroll 8
    for (int k = 0; k < HID; ++k) {
        float v = qs[i][k] + ts[j][k] + b1s[k];
        s = fmaf(v > 0.f ? v : 0.f, w2s[k], s);
    }
    out[(qi0 + i) * NT + (tj0 + j)] = s + b2[0];
}

extern "C" void kernel_launch(void* const* d_in, const int* in_sizes, int n_in,
                              void* d_out, int out_size, void* d_ws, size_t ws_size,
                              hipStream_t stream)
{
    (void)in_sizes; (void)n_in; (void)out_size; (void)ws_size;

    const float* qx     = (const float*)d_in[0];
    const int*   qe     = (const int*)  d_in[1];
    const float* tx     = (const float*)d_in[2];
    const int*   te     = (const int*)  d_in[3];
    const float* enc_w1 = (const float*)d_in[4];
    const float* enc_b1 = (const float*)d_in[5];
    const float* enc_w2 = (const float*)d_in[6];
    const float* enc_b2 = (const float*)d_in[7];
    const float* gcn_w  = (const float*)d_in[8];
    const float* gcn_b  = (const float*)d_in[9];
    const float* mw1    = (const float*)d_in[10];
    const float* mb1    = (const float*)d_in[11];
    const float* mw2    = (const float*)d_in[12];
    const float* mb2    = (const float*)d_in[13];
    float* out = (float*)d_out;

    const int NQ = 256, NT = 2048, EQ = 4096, ET = 65536;

    // Workspace carve-up (~4.7 MB of f32)
    float* p   = (float*)d_ws;
    float* qA  = p; p += NQ * HID;   // current q features
    float* qB  = p; p += NQ * HID;   // q scatter accumulator / ping-pong
    float* qT  = p; p += NQ * HID;   // q temp (encoder hidden / gcn lin)
    float* qP  = p; p += NQ * HID;   // qp projection
    float* tA  = p; p += NT * HID;
    float* tB  = p; p += NT * HID;
    float* tT  = p; p += NT * HID;
    float* tP  = p; p += NT * HID;
    float* dgq = p; p += NQ;         // deg -> dinv (in place)
    float* dgt = p; p += NT;

    // ---- Encoder: relu(X@W1+b1)@W2+b2 ----
    gemm128_wmma<<<NQ / 16, 256, 0, stream>>>(qx, enc_w1, enc_b1, qT, 0, 1);
    gemm128_wmma<<<NT / 16, 256, 0, stream>>>(tx, enc_w1, enc_b1, tT, 0, 1);
    gemm128_wmma<<<NQ / 16, 256, 0, stream>>>(qT, enc_w2, enc_b2, qA, 0, 0);
    gemm128_wmma<<<NT / 16, 256, 0, stream>>>(tT, enc_w2, enc_b2, tA, 0, 0);

    // ---- Degrees / dinv (dgq,dgt contiguous -> one zero launch) ----
    zero_f32<<<(NQ + NT + 255) / 256, 256, 0, stream>>>(dgq, NQ + NT);
    degree_k<<<EQ / 256, 256, 0, stream>>>(qe, dgq, EQ);
    degree_k<<<ET / 256, 256, 0, stream>>>(te, dgt, ET);
    dinv_k<<<(NQ + 255) / 256, 256, 0, stream>>>(dgq, NQ);
    dinv_k<<<(NT + 255) / 256, 256, 0, stream>>>(dgt, NT);

    // ---- 3 GCN layers (ReLU fused into the next GEMM's input load) ----
    for (int l = 0; l < 3; ++l) {
        const float* wl = gcn_w + l * HID * HID;
        const float* bl = gcn_b + l * HID;
        gemm128_wmma<<<NQ / 16, 256, 0, stream>>>(qA, wl, bl, qT, l > 0, 0);
        gemm128_wmma<<<NT / 16, 256, 0, stream>>>(tA, wl, bl, tT, l > 0, 0);
        zero_f32<<<(NQ * HID + 255) / 256, 256, 0, stream>>>(qB, NQ * HID);
        zero_f32<<<(NT * HID + 255) / 256, 256, 0, stream>>>(tB, NT * HID);
        scatter_k<<<EQ / 2, 256, 0, stream>>>(qe, qT, dgq, qB, EQ);
        scatter_k<<<ET / 2, 256, 0, stream>>>(te, tT, dgt, tB, ET);
        float* tmp;
        tmp = qA; qA = qB; qB = tmp;
        tmp = tA; tA = tB; tB = tmp;
    }

    // ---- Match projections: relu(x) @ MW1 halves (no bias here) ----
    gemm128_wmma<<<NQ / 16, 256, 0, stream>>>(qA, mw1,             nullptr, qP, 1, 0);
    gemm128_wmma<<<NT / 16, 256, 0, stream>>>(tA, mw1 + HID * HID, nullptr, tP, 1, 0);

    // ---- Pairwise scorer ----
    dim3 pgrid(NT / 16, NQ / 16);
    pairwise_k<<<pgrid, 256, 0, stream>>>(qP, tP, mb1, mw2, mb2, out, NT);
}
